// MultiHeadAttention_33011118637393
// MI455X (gfx1250) — matrix operand order repaired and checked by reading, not compiled, not run on hardware
//
#include <hip/hip_runtime.h>
#include <hip/hip_bf16.h>

#define HIDDEN 1024
#define HEADS  16
#define DH     64
#define BATCH  2
#define SEQ    2048
#define ROWS   (BATCH * SEQ)   // 4096

typedef __attribute__((ext_vector_type(16))) __bf16 v16bf;
typedef __attribute__((ext_vector_type(8)))  float  v8f;

// A-matrix (16x32 bf16) K index for vector element e in lane-half `half`:
// lanes 0-15: K=0..15 ; lanes 16-31: K=16..31 (contiguous, matches B)
__device__ __forceinline__ int a_kmap(int e, int half) {
    return 16 * half + e;
}

__device__ __forceinline__ v8f wmma_bf16(v16bf a, v16bf b, v8f c) {
    return __builtin_amdgcn_wmma_f32_16x16x32_bf16(
        false, a, false, b, (short)0, c, false, false);
}

// CDNA5 async global->LDS copy (16B per lane), tracked by ASYNCcnt.
__device__ __forceinline__ void async_copy_b128(void* lds_dst, const void* gsrc) {
    unsigned int  l = (unsigned int)(uintptr_t)lds_dst;           // LDS byte address
    unsigned long long g = (unsigned long long)(uintptr_t)gsrc;   // global address
    asm volatile("global_load_async_to_lds_b128 %0, %1, off"
                 :: "v"(l), "v"(g) : "memory");
}
__device__ __forceinline__ void wait_async0() {
    asm volatile("s_wait_asynccnt 0x0" ::: "memory");
}

// ---------------- stage 0: casts ----------------

__global__ void cast_f32_bf16(const float* __restrict__ in,
                              __bf16* __restrict__ out, int n) {
    int i = blockIdx.x * blockDim.x + threadIdx.x;
    if (i < n) out[i] = (__bf16)in[i];
}

// in: [HB, K, N] f32  ->  out: [HB, N, K] bf16
__global__ void castT_f32_bf16(const float* __restrict__ in,
                               __bf16* __restrict__ out, int HB, int K, int N) {
    int i = blockIdx.x * blockDim.x + threadIdx.x;
    int total = HB * K * N;
    if (i >= total) return;
    int n = i % N;
    int k = (i / N) % K;
    int h = i / (N * K);
    out[((size_t)h * N + n) * K + k] = (__bf16)in[i];
}

// ---------------- stage 1: fused QKV projection ----------------
// Block = 8 waves sharing (matrix, head); wave -> consecutive m-tiles.
// Weight tile [64 x 32k] staged via async->LDS, double-buffered.
__global__ __launch_bounds__(256)
void qkv_kernel(const __bf16* __restrict__ xb,    // [4096,1024]
                const __bf16* __restrict__ wqT,   // [16,64,1024]
                const __bf16* __restrict__ wkT,
                const __bf16* __restrict__ wvT,
                const float* __restrict__ bq,
                const float* __restrict__ bk,
                const float* __restrict__ bv,
                __bf16* __restrict__ q,           // [B*H, S, 64]
                __bf16* __restrict__ kout,        // [B*H, S, 64]
                __bf16* __restrict__ vt)          // [B*H, 64, S]
{
    __shared__ __bf16 wbuf[2][64][32];

    int tid  = threadIdx.x;
    int wid  = tid >> 5;
    int lane = tid & 31;
    int half = lane >> 4;
    int lm   = lane & 15;

    int w   = blockIdx.x * 8 + wid;
    int mat = w / (HEADS * (ROWS / 16));          // uniform per block
    int rem = w % (HEADS * (ROWS / 16));
    int h   = rem / (ROWS / 16);                  // uniform per block
    int mt  = rem % (ROWS / 16);

    const __bf16* wT   = (mat == 0) ? wqT : (mat == 1) ? wkT : wvT;
    const float*  bias = (mat == 0) ? bq  : (mat == 1) ? bk  : bv;
    const __bf16* wh   = wT + (size_t)h * DH * HIDDEN;

    // staging: 256 threads x 16B = 64 rows x 32 cols (bf16)
    int nrow  = tid >> 2;
    int chunk = tid & 3;

    v8f z = {};
    v8f acc[4];
#pragma unroll
    for (int t = 0; t < 4; t++) acc[t] = z;

    const __bf16* xrow = xb + (size_t)(mt * 16 + lm) * HIDDEN;

    async_copy_b128(&wbuf[0][nrow][chunk * 8],
                    wh + (size_t)nrow * HIDDEN + chunk * 8);

    for (int it = 0; it < HIDDEN / 32; ++it) {
        int k0  = it * 32;
        int buf = it & 1;
        wait_async0();
        __syncthreads();
        if (it + 1 < HIDDEN / 32)
            async_copy_b128(&wbuf[buf ^ 1][nrow][chunk * 8],
                            wh + (size_t)nrow * HIDDEN + (k0 + 32) + chunk * 8);

        v16bf a;
#pragma unroll
        for (int e = 0; e < 16; e++) a[e] = xrow[k0 + a_kmap(e, half)];
#pragma unroll
        for (int t = 0; t < 4; t++) {
            v16bf bf;
#pragma unroll
            for (int e = 0; e < 16; e++) bf[e] = wbuf[buf][t * 16 + lm][16 * half + e];
            acc[t] = wmma_bf16(a, bf, acc[t]);
        }
    }

#pragma unroll
    for (int t = 0; t < 4; t++) {
        float bsc = bias[h * DH + t * 16 + lm];
#pragma unroll
        for (int r = 0; r < 8; r++) {
            int m    = r + 8 * half;
            int grow = mt * 16 + m;
            int b    = grow >> 11;
            int s    = grow & (SEQ - 1);
            int d    = t * 16 + lm;
            int bh   = b * HEADS + h;
            __bf16 val = (__bf16)(acc[t][r] + bsc);
            if (mat == 0)      q[((size_t)bh * SEQ + s) * DH + d]    = val;
            else if (mat == 1) kout[((size_t)bh * SEQ + s) * DH + d] = val;
            else               vt[((size_t)bh * DH + d) * SEQ + s]   = val;
        }
    }
}

// ---------------- stage 2: flash attention ----------------
// Block = 8 waves sharing (b,h); wave -> 16-row Q block. K/V tiles (32 keys)
// staged via async->LDS, double-buffered; P re-swizzled through LDS.
__global__ __launch_bounds__(256)
void attn_kernel(const __bf16* __restrict__ q,    // [BH, S, 64]
                 const __bf16* __restrict__ kk,   // [BH, S, 64]
                 const __bf16* __restrict__ vt,   // [BH, 64, S]
                 __bf16* __restrict__ ctx)        // [B*S, 1024]
{
    __shared__ __bf16 kbuf[2][32][64];   // keys x d
    __shared__ __bf16 vbuf[2][64][32];   // d x keys (V transposed)
    __shared__ __bf16 lds_p[8][16][32];  // per-wave P tile

    int tid  = threadIdx.x;
    int wid  = tid >> 5;
    int lane = tid & 31;
    int half = lane >> 4;
    int lm   = lane & 15;

    int w  = blockIdx.x * 8 + wid;
    int bh = w >> 7;                      // uniform per block
    int qb = w & 127;
    int b  = bh >> 4, h = bh & 15;

    const __bf16* qbase = q  + (size_t)bh * SEQ * DH;
    const __bf16* kbase = kk + (size_t)bh * SEQ * DH;
    const __bf16* vbase = vt + (size_t)bh * DH * SEQ;

    // staging split: K tile 32x64 (8 chunks/row), V tile 64x32 (4 chunks/row)
    int krow = tid >> 3, kchunk = tid & 7;
    int vrow = tid >> 2, vchunk = tid & 3;

    // loop-invariant Q A-fragments
    v16bf qa[2];
    {
        const __bf16* qrow = qbase + (size_t)(qb * 16 + lm) * DH;
#pragma unroll
        for (int dk = 0; dk < 2; dk++)
#pragma unroll
            for (int e = 0; e < 16; e++)
                qa[dk][e] = qrow[dk * 32 + a_kmap(e, half)];
    }

    float m_i[8], l_i[8];
#pragma unroll
    for (int r = 0; r < 8; r++) { m_i[r] = -1e30f; l_i[r] = 0.0f; }
    v8f z = {};
    v8f o[4];
#pragma unroll
    for (int t = 0; t < 4; t++) o[t] = z;

    const float scale = 0.125f;

    async_copy_b128(&kbuf[0][krow][kchunk * 8],
                    kbase + (size_t)krow * DH + kchunk * 8);
    async_copy_b128(&vbuf[0][vrow][vchunk * 8],
                    vbase + (size_t)vrow * SEQ + vchunk * 8);

    for (int it = 0; it < SEQ / 32; ++it) {
        int kb  = it * 32;
        int buf = it & 1;
        wait_async0();
        __syncthreads();
        if (it + 1 < SEQ / 32) {
            int kb2 = kb + 32;
            async_copy_b128(&kbuf[buf ^ 1][krow][kchunk * 8],
                            kbase + (size_t)(kb2 + krow) * DH + kchunk * 8);
            async_copy_b128(&vbuf[buf ^ 1][vrow][vchunk * 8],
                            vbase + (size_t)vrow * SEQ + kb2 + vchunk * 8);
        }

        // scores: two 16x16 key tiles, 2 WMMAs each over d
        v8f s0 = z, s1 = z;
#pragma unroll
        for (int dk = 0; dk < 2; dk++) {
            v16bf b0, b1;
#pragma unroll
            for (int e = 0; e < 16; e++) {
                b0[e] = kbuf[buf][lm][dk * 32 + 16 * half + e];
                b1[e] = kbuf[buf][16 + lm][dk * 32 + 16 * half + e];
            }
            s0 = wmma_bf16(qa[dk], b0, s0);
            s1 = wmma_bf16(qa[dk], b1, s1);
        }

        // online softmax; row (r,half) lives in one 16-lane half-wave
#pragma unroll
        for (int r = 0; r < 8; r++) {
            float sa = s0[r] * scale, sb = s1[r] * scale;
            float mloc = fmaxf(sa, sb);
#pragma unroll
            for (int off = 1; off < 16; off <<= 1)
                mloc = fmaxf(mloc, __shfl_xor(mloc, off, 32));
            float mn = fmaxf(m_i[r], mloc);
            float al = __expf(m_i[r] - mn);
            float p0 = __expf(sa - mn);
            float p1 = __expf(sb - mn);
            float rs = p0 + p1;
#pragma unroll
            for (int off = 1; off < 16; off <<= 1)
                rs += __shfl_xor(rs, off, 32);
            l_i[r] = l_i[r] * al + rs;
            m_i[r] = mn;
#pragma unroll
            for (int t = 0; t < 4; t++) o[t][r] *= al;
            lds_p[wid][r + 8 * half][lm]      = (__bf16)p0;
            lds_p[wid][r + 8 * half][16 + lm] = (__bf16)p1;
        }
        __syncthreads();
        v16bf pa;
#pragma unroll
        for (int e = 0; e < 16; e++) pa[e] = lds_p[wid][lm][a_kmap(e, half)];
        __syncthreads();

        // ctx += P(16x32) @ V(32x64)
#pragma unroll
        for (int t = 0; t < 4; t++) {
            v16bf vb;
#pragma unroll
            for (int e = 0; e < 16; e++)
                vb[e] = vbuf[buf][t * 16 + lm][16 * half + e];
            o[t] = wmma_bf16(pa, vb, o[t]);
        }
    }

#pragma unroll
    for (int r = 0; r < 8; r++) {
        float inv = 1.0f / l_i[r];
        int m    = r + 8 * half;
        int srow = qb * 16 + m;
        size_t orow = ((size_t)b * SEQ + srow) * HIDDEN + h * DH;
#pragma unroll
        for (int t = 0; t < 4; t++)
            ctx[orow + t * 16 + lm] = (__bf16)(o[t][r] * inv);
    }
}

// ---------------- stage 3: output projection ----------------
// Block = 8 waves sharing an n-strip of Wo; Wo tile staged via async->LDS.
__global__ __launch_bounds__(256)
void oproj_kernel(const __bf16* __restrict__ ctxb,  // [4096,1024]
                  const __bf16* __restrict__ woT,   // [n=1024][k=1024]
                  const float* __restrict__ bo,
                  float* __restrict__ out)          // [4096,1024] f32
{
    __shared__ __bf16 wbuf[2][64][32];

    int tid  = threadIdx.x;
    int wid  = tid >> 5;
    int lane = tid & 31;
    int half = lane >> 4;
    int lm   = lane & 15;

    int w  = blockIdx.x * 8 + wid;
    int ns = w >> 8;          // n-strip (uniform per block)
    int mt = w & 255;         // m-tile

    const __bf16* wbase = woT + (size_t)ns * 64 * HIDDEN;

    int nrow  = tid >> 2;
    int chunk = tid & 3;

    v8f z = {};
    v8f acc[4];
#pragma unroll
    for (int t = 0; t < 4; t++) acc[t] = z;

    const __bf16* arow = ctxb + (size_t)(mt * 16 + lm) * HIDDEN;

    async_copy_b128(&wbuf[0][nrow][chunk * 8],
                    wbase + (size_t)nrow * HIDDEN + chunk * 8);

    for (int it = 0; it < HIDDEN / 32; ++it) {
        int k0  = it * 32;
        int buf = it & 1;
        wait_async0();
        __syncthreads();
        if (it + 1 < HIDDEN / 32)
            async_copy_b128(&wbuf[buf ^ 1][nrow][chunk * 8],
                            wbase + (size_t)nrow * HIDDEN + (k0 + 32) + chunk * 8);

        v16bf a;
#pragma unroll
        for (int e = 0; e < 16; e++) a[e] = arow[k0 + a_kmap(e, half)];
#pragma unroll
        for (int t = 0; t < 4; t++) {
            v16bf bf;
#pragma unroll
            for (int e = 0; e < 16; e++) bf[e] = wbuf[buf][t * 16 + lm][16 * half + e];
            acc[t] = wmma_bf16(a, bf, acc[t]);
        }
    }

#pragma unroll
    for (int t = 0; t < 4; t++) {
        float bsc = bo[ns * 64 + t * 16 + lm];
#pragma unroll
        for (int r = 0; r < 8; r++) {
            int m = r + 8 * half;
            out[((size_t)(mt * 16 + m)) * HIDDEN + ns * 64 + t * 16 + lm] = acc[t][r] + bsc;
        }
    }
}

// ---------------- host launch ----------------
extern "C" void kernel_launch(void* const* d_in, const int* in_sizes, int n_in,
                              void* d_out, int out_size, void* d_ws, size_t ws_size,
                              hipStream_t stream) {
    const float* x  = (const float*)d_in[0];
    const float* Wq = (const float*)d_in[1];
    const float* bq = (const float*)d_in[2];
    const float* Wk = (const float*)d_in[3];
    const float* bk = (const float*)d_in[4];
    const float* Wv = (const float*)d_in[5];
    const float* bv = (const float*)d_in[6];
    const float* Wo = (const float*)d_in[7];
    const float* bo = (const float*)d_in[8];
    float* out = (float*)d_out;

    char* p = (char*)d_ws;
    size_t off = 0;
    auto take = [&](size_t bytes) {
        char* r = p + off;
        off += (bytes + 255) & ~(size_t)255;
        return r;
    };
    __bf16* xb  = (__bf16*)take((size_t)ROWS * HIDDEN * 2);
    __bf16* wqT = (__bf16*)take((size_t)HEADS * DH * HIDDEN * 2);
    __bf16* wkT = (__bf16*)take((size_t)HEADS * DH * HIDDEN * 2);
    __bf16* wvT = (__bf16*)take((size_t)HEADS * DH * HIDDEN * 2);
    __bf16* woT = (__bf16*)take((size_t)HIDDEN * HIDDEN * 2);
    __bf16* qw  = (__bf16*)take((size_t)BATCH * HEADS * SEQ * DH * 2);
    __bf16* kw  = (__bf16*)take((size_t)BATCH * HEADS * SEQ * DH * 2);
    __bf16* vtw = (__bf16*)take((size_t)BATCH * HEADS * DH * SEQ * 2);
    __bf16* ctx = (__bf16*)take((size_t)ROWS * HIDDEN * 2);
    (void)ws_size; (void)n_in; (void)in_sizes; (void)out_size;

    {
        int n = ROWS * HIDDEN;
        cast_f32_bf16<<<(n + 255) / 256, 256, 0, stream>>>(x, xb, n);
    }
    {
        int n = HEADS * HIDDEN * DH;
        castT_f32_bf16<<<(n + 255) / 256, 256, 0, stream>>>(Wq, wqT, HEADS, HIDDEN, DH);
        castT_f32_bf16<<<(n + 255) / 256, 256, 0, stream>>>(Wk, wkT, HEADS, HIDDEN, DH);
        castT_f32_bf16<<<(n + 255) / 256, 256, 0, stream>>>(Wv, wvT, HEADS, HIDDEN, DH);
    }
    {
        int n = HIDDEN * HIDDEN;
        castT_f32_bf16<<<(n + 255) / 256, 256, 0, stream>>>(Wo, woT, 1, HIDDEN, HIDDEN);
    }

    qkv_kernel<<<1536, 256, 0, stream>>>(xb, wqT, wkT, wvT, bq, bk, bv, qw, kw, vtw);
    attn_kernel<<<512, 256, 0, stream>>>(qw, kw, vtw, ctx);
    oproj_kernel<<<512, 256, 0, stream>>>(ctx, woT, bo, out);
}
